// LMUModel_20710332301583
// MI455X (gfx1250) — matrix operand order repaired and checked by reading, not compiled, not run on hardware
//
#include <hip/hip_runtime.h>

// ---------------------------------------------------------------------------
// LMU forward scan for gfx1250 (MI455X), round 5.
//  - 16 workgroups x 16 waves; each WG owns a 16-row batch tile.
//  - State (h, m) in LDS as bf16 in exact WMMA A-fragment order
//    (A-operand build = two ds_load_b128, zero VALU).
//  - Weights pre-packed into exact B-fragment order (coalesced L2 streams).
//  - Partial unroll (no spills); branch-free prefetch bursts per phase;
//    4 independent WMMA accumulator chains per phase.
// ---------------------------------------------------------------------------

typedef __attribute__((ext_vector_type(16))) __bf16          v16bf;
typedef __attribute__((ext_vector_type(16))) unsigned short  v16u;
typedef __attribute__((ext_vector_type(8)))  float           v8f;

#define L_HID   512
#define L_MEM   512
#define L_SEQ   784
#define L_BATCH 256
#define L_OUT   10
#define TILE_B  16
#define NTHREADS 512          // 16 waves (wave32)
#define KC_STEPS 16           // 512 / 32 K-chunks
#define KC_STRIDE 512         // ushorts per K-chunk fragment block (1 KB)
#define NT_STRIDE 8192        // ushorts per 16-wide N tile in packed weights
#define FRAG_ELEMS (16 * 512) // one state tile in fragment order (ushorts)

__device__ __forceinline__ unsigned short f2bf(float f) {
    unsigned int u = __float_as_uint(f);
    u += 0x7FFFu + ((u >> 16) & 1u);      // round-to-nearest-even
    return (unsigned short)(u >> 16);
}
__device__ __forceinline__ float bf2f(unsigned short b) {
    return __uint_as_float(((unsigned int)b) << 16);
}
__device__ __forceinline__ v16bf ldfrag(const unsigned short* p) {
    return __builtin_bit_cast(v16bf, *(const v16u*)p);
}

// A-fragment index for state element (row M, col k); matches ISA 16x32 layout.
__device__ __forceinline__ int frag_idx(int M, int k) {
    int kc   = k >> 5;
    int K    = k & 31;
    int half = (K >> 3) & 1;
    int hi   = K >> 4;
    int e    = 2 * (hi * 4 + ((K & 7) >> 1)) + (K & 1);
    return ((kc * 32 + half * 16 + M) << 4) + e;
}

// Branch-free prefetch burst: every other K-chunk of one weight stream.
__device__ __forceinline__ void prefetch_stream(const unsigned short* base) {
#pragma unroll
    for (int pk = 0; pk < KC_STEPS; pk += 2)
        __builtin_prefetch(base + pk * KC_STRIDE, 0, 3);
}

// Pack W[512][512] (row-major f32; row = N, col = K) into bf16 B-fragment
// order: dst[nt*8192 + kc*512 + lane*16 + e] = W[nt*16+lane%16][kc*32+K],
// K = 8*(lane/16)+e (e<8), 16+8*(lane/16)+(e-8) (e>=8).
__global__ void lmu_pack_kernel(const float* __restrict__ W,
                                unsigned short* __restrict__ dst) {
    int idx = blockIdx.x * blockDim.x + threadIdx.x;
    if (idx >= L_HID * L_HID) return;
    int i = idx >> 9;          // N
    int j = idx & 511;         // K
    int nt = i >> 4, kc = j >> 5, K = j & 31;
    int lane = (((K >> 3) & 1) << 4) + (i & 15), e = ((K >> 4) << 3) + (K & 7);
    dst[nt * NT_STRIDE + kc * KC_STRIDE + lane * 16 + e] = f2bf(W[idx]);
}

__global__ __launch_bounds__(NTHREADS, 1)
void lmu_scan_kernel(const float* __restrict__ x,
                     const float* __restrict__ e_x,
                     const float* __restrict__ e_h,
                     const float* __restrict__ e_m,
                     const float* __restrict__ W_x,
                     const float* __restrict__ Bd,
                     const float* __restrict__ W_cls,
                     const float* __restrict__ b_cls,
                     const unsigned short* __restrict__ pA,
                     const unsigned short* __restrict__ pWh,
                     const unsigned short* __restrict__ pWm,
                     float* __restrict__ out) {
    extern __shared__ unsigned char smraw[];
    unsigned short* mb0 = (unsigned short*)smraw;            // [8192] bf16, frag order
    unsigned short* mb1 = mb0 + FRAG_ELEMS;
    unsigned short* hb0 = mb1 + FRAG_ELEMS;
    unsigned short* hb1 = hb0 + FRAG_ELEMS;
    float* sehp = (float*)(hb1 + FRAG_ELEMS);                // [512] e_h permuted
    float* semp = sehp + L_HID;                              // [512] e_m permuted
    float* sB   = semp + L_MEM;                              // [512] natural order
    float* sWx  = sB + L_MEM;                                // [512] natural order
    float* sred = sWx + L_HID;                               // [16][32]
    float* su   = sred + TILE_B * 32;                        // [16]
    float* sx   = su + TILE_B;                               // [16]

    const int tid  = threadIdx.x;
    const int b0   = blockIdx.x * TILE_B;
    const int wave = tid >> 5;
    const int lane = tid & 31;
    const int half = lane >> 4;
    const int rr   = lane & 15;

    // zero bf16 state; load + permute small vectors
    for (int i = tid; i < FRAG_ELEMS; i += NTHREADS) { mb0[i] = 0; hb0[i] = 0; }
    if (tid < L_HID) {
        int k = tid;
        int kc = k >> 5, K = k & 31;
        int hf = (K >> 3) & 1, hi = K >> 4;
        int e  = 2 * (hi * 4 + ((K & 7) >> 1)) + (K & 1);
        int ci = (((kc << 1) + hf) << 4) + e;
        sehp[ci] = e_h[k];
        semp[ci] = e_m[k];
        sB[k]  = Bd[k];
        sWx[k] = W_x[k];
    }
    __syncthreads();

    const float ex0 = e_x[0];
    const int nt0 = wave * 2, nt1 = wave * 2 + 1;
    const int n0 = nt0 * 16 + rr, n1 = nt1 * 16 + rr;
    const int fb0 = frag_idx(0, n0);       // + M*16 gives element address
    const int fb1 = frag_idx(0, n1);

    // hoisted per-wave base pointers (hot-loop addrs = base + const offset)
    const unsigned short* bA  = pA  + (size_t)nt0 * NT_STRIDE + lane * 16; // nt1 = +NT_STRIDE
    const unsigned short* bWh = pWh + (size_t)nt0 * NT_STRIDE + lane * 16;
    const unsigned short* bWm = pWm + (size_t)nt0 * NT_STRIDE + lane * 16;

    unsigned short *mc = mb0, *mn = mb1, *hc = hb0, *hn = hb1;
    const int laneoff = lane * 16;

    for (int t = 0; t < L_SEQ; ++t) {
        // ---- phase 1: u[b] = e_x*x[b,t] + h[b].e_h + m[b].e_m ----
        {
            int r = tid >> 5, c0 = tid & 31;
            int hf = c0 >> 4, e = c0 & 15;
            float p = 0.f;
#pragma unroll 4
            for (int kc = 0; kc < KC_STEPS; ++kc) {
                int i  = ((kc * 32 + hf * 16 + r) << 4) + e;
                int ci = (((kc << 1) + hf) << 4) + e;
                p += bf2f(hc[i]) * sehp[ci] + bf2f(mc[i]) * semp[ci];
            }
            sred[r * 32 + c0] = p;
        }
        __syncthreads();
        if (tid < TILE_B) {
            float s = 0.f;
#pragma unroll
            for (int j = 0; j < 32; ++j) s += sred[tid * 32 + j];
            float xv = x[(size_t)(b0 + tid) * L_SEQ + t];
            sx[tid] = xv;
            su[tid] = s + ex0 * xv;
        }
        __syncthreads();

        // ---- phase 2: m_new = m @ A^T + u * B^T  (4 chains, K split) ----
        {
            prefetch_stream(bA);
            prefetch_stream(bA + NT_STRIDE);
            const unsigned short* mcl = mc + laneoff;
            v8f a0a = {}, a1a = {}, a0b = {}, a1b = {};
#pragma unroll 2
            for (int kc = 0; kc < KC_STEPS / 2; ++kc) {
                const int oa = kc * KC_STRIDE;
                const int ob = (kc + 8) * KC_STRIDE;
                v16bf aA = ldfrag(mcl + oa);
                v16bf aB = ldfrag(mcl + ob);
                a0a = __builtin_amdgcn_wmma_f32_16x16x32_bf16(false, aA, false, ldfrag(bA + oa), (short)0, a0a, false, false);
                a1a = __builtin_amdgcn_wmma_f32_16x16x32_bf16(false, aA, false, ldfrag(bA + NT_STRIDE + oa), (short)0, a1a, false, false);
                a0b = __builtin_amdgcn_wmma_f32_16x16x32_bf16(false, aB, false, ldfrag(bA + ob), (short)0, a0b, false, false);
                a1b = __builtin_amdgcn_wmma_f32_16x16x32_bf16(false, aB, false, ldfrag(bA + NT_STRIDE + ob), (short)0, a1b, false, false);
            }
#pragma unroll
            for (int v = 0; v < 8; ++v) {
                int M = v + 8 * half;           // D layout: M = v + 8*half
                mn[fb0 + (M << 4)] = f2bf((a0a[v] + a0b[v]) + su[M] * sB[n0]);
                mn[fb1 + (M << 4)] = f2bf((a1a[v] + a1b[v]) + su[M] * sB[n1]);
            }
        }
        __syncthreads();

        // ---- phase 3: h_new = tanh(x*Wx + h@Wh^T + m_new@Wm^T), fused ----
        {
            prefetch_stream(bWh);
            prefetch_stream(bWh + NT_STRIDE);
            prefetch_stream(bWm);
            prefetch_stream(bWm + NT_STRIDE);
            const unsigned short* hcl = hc + laneoff;
            const unsigned short* mnl = mn + laneoff;
            v8f ah0 = {}, ah1 = {}, am0 = {}, am1 = {};
#pragma unroll 2
            for (int kc = 0; kc < KC_STEPS; ++kc) {
                const int o = kc * KC_STRIDE;
                v16bf ah = ldfrag(hcl + o);
                v16bf am = ldfrag(mnl + o);
                ah0 = __builtin_amdgcn_wmma_f32_16x16x32_bf16(false, ah, false, ldfrag(bWh + o), (short)0, ah0, false, false);
                ah1 = __builtin_amdgcn_wmma_f32_16x16x32_bf16(false, ah, false, ldfrag(bWh + NT_STRIDE + o), (short)0, ah1, false, false);
                am0 = __builtin_amdgcn_wmma_f32_16x16x32_bf16(false, am, false, ldfrag(bWm + o), (short)0, am0, false, false);
                am1 = __builtin_amdgcn_wmma_f32_16x16x32_bf16(false, am, false, ldfrag(bWm + NT_STRIDE + o), (short)0, am1, false, false);
            }
#pragma unroll
            for (int v = 0; v < 8; ++v) {
                int M = v + 8 * half;
                hn[fb0 + (M << 4)] = f2bf(tanhf((ah0[v] + am0[v]) + sx[M] * sWx[n0]));
                hn[fb1 + (M << 4)] = f2bf(tanhf((ah1[v] + am1[v]) + sx[M] * sWx[n1]));
            }
        }
        __syncthreads();

        unsigned short* tp = mc; mc = mn; mn = tp;
        tp = hc; hc = hn; hn = tp;
    }

    // ---- classifier: out = h_n @ W_cls^T + b_cls ----
    for (int idx = tid; idx < TILE_B * L_OUT; idx += NTHREADS) {
        int r = idx / L_OUT, o = idx - r * L_OUT;
        float s = b_cls[o];
        for (int c = 0; c < L_HID; ++c)
            s += bf2f(hc[frag_idx(r, c)]) * W_cls[o * L_HID + c];
        out[(size_t)(b0 + r) * L_OUT + o] = s;
    }
}

extern "C" void kernel_launch(void* const* d_in, const int* in_sizes, int n_in,
                              void* d_out, int out_size, void* d_ws, size_t ws_size,
                              hipStream_t stream) {
    (void)in_sizes; (void)n_in; (void)out_size; (void)ws_size;
    const float* x     = (const float*)d_in[0];
    const float* e_x   = (const float*)d_in[1];
    const float* e_h   = (const float*)d_in[2];
    const float* e_m   = (const float*)d_in[3];
    const float* W_x   = (const float*)d_in[4];
    const float* W_h   = (const float*)d_in[5];
    const float* W_m   = (const float*)d_in[6];
    const float* A     = (const float*)d_in[7];
    const float* Bd    = (const float*)d_in[8];
    const float* W_cls = (const float*)d_in[9];
    const float* b_cls = (const float*)d_in[10];
    float* out = (float*)d_out;

    unsigned short* pA  = (unsigned short*)d_ws;
    unsigned short* pWh = pA  + L_HID * L_HID;
    unsigned short* pWm = pWh + L_HID * L_HID;

    const int n = L_HID * L_HID;
    lmu_pack_kernel<<<dim3((n + 255) / 256), dim3(256), 0, stream>>>(A,   pA);
    lmu_pack_kernel<<<dim3((n + 255) / 256), dim3(256), 0, stream>>>(W_h, pWh);
    lmu_pack_kernel<<<dim3((n + 255) / 256), dim3(256), 0, stream>>>(W_m, pWm);

    const size_t shmem = (size_t)(4 * FRAG_ELEMS) * sizeof(unsigned short) +
                         (size_t)(4 * L_HID + TILE_B * 32 + 2 * TILE_B) * sizeof(float);
    lmu_scan_kernel<<<dim3(L_BATCH / TILE_B), dim3(NTHREADS), shmem, stream>>>(
        x, e_x, e_h, e_m, W_x, Bd, W_cls, b_cls, pA, pWh, pWm, out);
}
